// CostVolumeLayer_60129542873
// MI455X (gfx1250) — matrix operand order repaired and checked by reading, not compiled, not run on hardware
//
#include <hip/hip_runtime.h>

// CDNA5 (gfx1250) PWC-Net cost volume via v_wmma_f32_16x16x32_f16.
//
// Pass 1: convert+zero-pad x2 fp32 -> f16 into workspace [8][136][144][128]
//         (row pad 4 each side, col pad 8 each side -> 144).
// Pass 2: per (b,h) block, 8 waves, wave w owns 16-pixel tile w0=16w.
//         For each vertical shift di in [-4,4]: banded matmul
//         out[w, w'] = sum_c x1[h,w,c] * x2h[h-di, w', c]
//         over a 32-wide x2 strip (two 16x16 N-tiles, K=128 via 4 chained
//         WMMAs), then extract the 9 diagonals through private LDS and
//         scatter to out[b,h,w, (9*di+dj) mod 81] * (1/128).
//
// Roofline: 2.7 GFLOP useful / ~170 MB compulsory HBM traffic -> HBM-bound
// (~7-10 us) once the dot products run on the f16 matrix pipe; x2h (38 MiB)
// stays resident in the 192 MB L2 for its 18x reuse.

typedef __attribute__((ext_vector_type(16))) _Float16 v16h;
typedef __attribute__((ext_vector_type(4)))  _Float16 v4h;
typedef __attribute__((ext_vector_type(8)))  float    v8f;

#define Bb 8
#define Hh 128
#define Ww 128
#define Cc 128
#define HP 136   // 128 + 2*4
#define WP 144   // 128 + 2*8
#define NCH 81

__global__ __launch_bounds__(256) void cv_pad_cvt_x2(
    const float* __restrict__ x2, _Float16* __restrict__ x2h) {
  // one thread = 4 channels of one padded pixel
  long tid = (long)blockIdx.x * blockDim.x + threadIdx.x;
  const long total = (long)Bb * HP * WP * (Cc / 4);
  if (tid >= total) return;
  int cg  = (int)(tid & 31);          // channel group (4 ch each)
  long pix = tid >> 5;                // [0, B*HP*WP)
  int wp = (int)(pix % WP);
  long t  = pix / WP;
  int hp = (int)(t % HP);
  int b  = (int)(t / HP);
  _Float16* dst = x2h + pix * Cc + cg * 4;
  int h = hp - 4, w = wp - 8;
  v4h o;
  if (h >= 0 && h < Hh && w >= 0 && w < Ww) {
    const float* src = x2 + ((((long)b * Hh + h) * Ww + w) * Cc) + cg * 4;
    float4 f = *(const float4*)src;
    o[0] = (_Float16)f.x; o[1] = (_Float16)f.y;
    o[2] = (_Float16)f.z; o[3] = (_Float16)f.w;
  } else {
    o = (v4h)0;
  }
  *(v4h*)dst = o;
}

__global__ __launch_bounds__(256) void cv_wmma(
    const float* __restrict__ x1, const _Float16* __restrict__ x2h,
    float* __restrict__ out) {
  __shared__ float lds[8][16 * 32];          // private 16x32 f32 tile per wave

  const int lane = threadIdx.x & 31;
  const int wave = threadIdx.x >> 5;
  const int bh = blockIdx.x;                 // [0, B*H)
  const int h = bh & (Hh - 1);
  const int b = bh >> 7;
  const int w0 = wave * 16;

  // ---- A fragments: x1[b,h, w0+(lane%16), :] fp32 -> f16 on the fly ----
  // v16h layout (16-bit A 16x32): halves 0..15 = ch c0+s .. c0+s+15
  // with s = (lane<16 ? 0 : 16), row M = lane%16.
  const int m16 = lane & 15;
  const int s = (lane < 16) ? 0 : 16;
  const float* x1p = x1 + (((long)b * Hh + h) * Ww + (w0 + m16)) * Cc;
  v16h afrag[4];
  #pragma unroll
  for (int kf = 0; kf < 4; ++kf) {
    const int c0 = kf * 32;
    float4 f0 = *(const float4*)(x1p + c0 + s);
    float4 f1 = *(const float4*)(x1p + c0 + s + 4);
    float4 f2 = *(const float4*)(x1p + c0 + s + 8);
    float4 f3 = *(const float4*)(x1p + c0 + s + 12);
    float tf[16];
    *(float4*)(tf + 0)  = f0; *(float4*)(tf + 4)  = f1;
    *(float4*)(tf + 8)  = f2; *(float4*)(tf + 12) = f3;
    v16h a;
    #pragma unroll
    for (int e = 0; e < 16; ++e) a[e] = (_Float16)tf[e];
    afrag[kf] = a;
  }

  const float scale = 1.0f / (float)Hh;
  float* outp = out + (((long)b * Hh + h) * Ww) * NCH;
  float* ldsw = lds[wave];
  const int n = lane & 15;                   // B/D column within tile
  const int cs = (lane < 16) ? 0 : 16;       // B K-half select
  const int mbase = (lane < 16) ? 0 : 8;     // D row base

  // ---- Hoist diagonal-extraction metadata (invariant across di) ----
  // Flat element e = lane + 32k over 16 rows x 9 shifts; mm = e/9,
  // dj = e%9 - 4, strip column = mm + 8 - dj.
  int ldsoffv[5], outoffv[5], djv[5];
  #pragma unroll
  for (int k = 0; k < 5; ++k) {
    const int e = lane + 32 * k;
    const int ee = (e < 144) ? e : 0;        // keep indices sane for dead lanes
    const int mm = ee / 9;
    const int dj = ee % 9 - 4;
    djv[k] = dj;
    ldsoffv[k] = mm * 32 + (mm + 8 - dj);
    outoffv[k] = (w0 + mm) * NCH;
  }

  for (int i = 0; i < 9; ++i) {
    const int di = i - 4;
    const int r = h - di + 4;                // padded x2 row, in [0,136)
    const _Float16* rowp = x2h + ((long)b * HP + r) * (long)WP * Cc;

    v8f acc0 = {};
    v8f acc1 = {};
    #pragma unroll
    for (int kf = 0; kf < 4; ++kf) {
      const int c0 = kf * 32 + cs;
      // B fragment: 16 contiguous f16 channels of pixel (padded col)
      v16h b0 = *(const v16h*)(rowp + (long)(w0 + n) * Cc + c0);
      v16h b1 = *(const v16h*)(rowp + (long)(w0 + 16 + n) * Cc + c0);
      acc0 = __builtin_amdgcn_wmma_f32_16x16x32_f16(
          false, afrag[kf], false, b0, (short)0, acc0, false, false);
      acc1 = __builtin_amdgcn_wmma_f32_16x16x32_f16(
          false, afrag[kf], false, b1, (short)0, acc1, false, false);
    }

    // Stage 16x32 result in this wave's private LDS tile.
    #pragma unroll
    for (int rr = 0; rr < 8; ++rr) {
      ldsw[(mbase + rr) * 32 + n]      = acc0[rr];
      ldsw[(mbase + rr) * 32 + 16 + n] = acc1[rr];
    }
    // Same-wave DS ops are in-order: loads below see the stores above.

    // Extract the 9 diagonals; ch = (9*di + dj) mod 81, 9*di+dj in [-40,40]
    // so the torch wraparound is a single conditional +81.
    const int base9 = 9 * di;
    #pragma unroll
    for (int k = 0; k < 5; ++k) {
      if (lane + 32 * k < 144) {
        int ch = base9 + djv[k];
        ch += (ch < 0) ? NCH : 0;
        outp[outoffv[k] + ch] = ldsw[ldsoffv[k]] * scale;
      }
    }
  }
}

extern "C" void kernel_launch(void* const* d_in, const int* in_sizes, int n_in,
                              void* d_out, int out_size, void* d_ws, size_t ws_size,
                              hipStream_t stream) {
  const float* x1 = (const float*)d_in[0];
  const float* x2 = (const float*)d_in[1];
  float* outp = (float*)d_out;
  // workspace: padded f16 x2, 8*136*144*128*2 = 40,108,032 bytes
  _Float16* x2h = (_Float16*)d_ws;

  const long total_p1 = (long)Bb * HP * WP * (Cc / 4);
  const int blk = 256;
  const int grid_p1 = (int)((total_p1 + blk - 1) / blk);
  cv_pad_cvt_x2<<<grid_p1, blk, 0, stream>>>(x2, x2h);

  cv_wmma<<<Bb * Hh, 256, 0, stream>>>(x1, x2h, outp);
}
